// mymodel_22041772163407
// MI455X (gfx1250) — matrix operand order repaired and checked by reading, not compiled, not run on hardware
//
#include <hip/hip_runtime.h>
#include <hip/hip_bf16.h>

// MCConv point-cloud network for MI455X (gfx1250, wave32, WMMA).
// GEMM-dominant workload -> bf16 WMMA (v_wmma_f32_16x16x32_bf16) with f32
// accumulate; f32->bf16 conversion fused into LDS tile staging; W staged
// TRANSPOSED in LDS so B fragments are contiguous (ds_load_b128, no b16 picks).

#define ALPHA   0.3f
#define BN_INV  0.9995003746871094f   // 1/sqrt(1.001)

typedef __attribute__((ext_vector_type(16))) __bf16 v16bf;
typedef __attribute__((ext_vector_type(8)))  float  v8f;
typedef __attribute__((ext_vector_type(4)))  float  f4;
typedef __attribute__((ext_vector_type(4)))  __bf16 bf4;

__device__ __forceinline__ float lrelu(float v) { return v >= 0.0f ? v : ALPHA * v; }

// ---------------------------------------------------------------------------
// 1. KNN within radius: per-query K nearest sources (top_k(-d2) semantics).
//    idx = -1 marks invalid (outside radius / fewer than K in range).
// ---------------------------------------------------------------------------
__global__ void mc_knn_kernel(const float* __restrict__ pts,   // [16,4096,3]
                              int S, int Q, int K, float r2,
                              int* __restrict__ idx,           // [16*Q*K]
                              float* __restrict__ cnt)         // [16*Q]
{
    int gid = blockIdx.x * blockDim.x + threadIdx.x;
    int total = 16 * Q;
    if (gid >= total) return;
    int b = gid / Q, q = gid % Q;
    const float* pb = pts + (size_t)b * 4096 * 3;
    float qx = pb[q * 3 + 0], qy = pb[q * 3 + 1], qz = pb[q * 3 + 2];

    float bd[64];
    int   bi[64];
    int   m = 0;         // filled entries
    int   wi = 0;        // index of current worst (largest d2)
    float wd = -1.0f;

    for (int s = 0; s < S; ++s) {
        float dx = pb[s * 3 + 0] - qx;
        float dy = pb[s * 3 + 1] - qy;
        float dz = pb[s * 3 + 2] - qz;
        float d2 = dx * dx + dy * dy + dz * dz;
        if (d2 > r2) continue;
        if (m < K) {
            bd[m] = d2; bi[m] = s;
            if (d2 > wd) { wd = d2; wi = m; }
            ++m;
        } else if (d2 < wd) {
            bd[wi] = d2; bi[wi] = s;
            wd = -1.0f;
            for (int j = 0; j < K; ++j)
                if (bd[j] > wd) { wd = bd[j]; wi = j; }
        }
    }
    int* op = idx + (size_t)gid * K;
    for (int j = 0; j < K; ++j) op[j] = (j < m) ? bi[j] : -1;
    cnt[gid] = (float)(m > 0 ? m : 1);
}

// ---------------------------------------------------------------------------
// 2. Basis MLP + neighborhood aggregation. One workgroup per (b,q).
// ---------------------------------------------------------------------------
__global__ void mc_agg_kernel(const float* __restrict__ pts,
                              const float* __restrict__ fsrc,   // [16,S,cin]
                              int cin, int S, int Q, int K, float rinv,
                              const float* __restrict__ wb,     // [3,32]
                              const float* __restrict__ bb,     // [32]
                              const int* __restrict__ idx,
                              const float* __restrict__ cnt,
                              float* __restrict__ agg)          // [16*Q, 32*cin]
{
    __shared__ float s_wb[96];
    __shared__ float s_bb[32];
    __shared__ int   s_idx[64];
    __shared__ float s_basis[64 * 33];

    int t = threadIdx.x;
    int b = blockIdx.x / Q, q = blockIdx.x % Q;

    if (t < 96)                s_wb[t]      = wb[t];
    else if (t < 128)          s_bb[t - 96] = bb[t - 96];
    if (t < K)                 s_idx[t] = idx[(size_t)blockIdx.x * K + t];
    __syncthreads();

    if (t < K) {
        float* br = &s_basis[t * 33];
        int j = s_idx[t];
        if (j >= 0) {
            const float* pb = pts + (size_t)b * 4096 * 3;
            float rx = (pb[j * 3 + 0] - pb[q * 3 + 0]) * rinv;
            float ry = (pb[j * 3 + 1] - pb[q * 3 + 1]) * rinv;
            float rz = (pb[j * 3 + 2] - pb[q * 3 + 2]) * rinv;
            #pragma unroll
            for (int h = 0; h < 32; ++h) {
                float v = rx * s_wb[h] + ry * s_wb[32 + h] + rz * s_wb[64 + h] + s_bb[h];
                br[h] = lrelu(v);
            }
        } else {
            #pragma unroll
            for (int h = 0; h < 32; ++h) br[h] = 0.0f;
        }
    }
    __syncthreads();

    float invc = 1.0f / cnt[blockIdx.x];
    int HC = 32 * cin;
    for (int e = t; e < HC; e += 256) {
        int h = e / cin, c = e - h * cin;
        float a = 0.0f;
        for (int k = 0; k < K; ++k) {
            int j = s_idx[k];
            if (j >= 0)
                a += s_basis[k * 33 + h] * fsrc[((size_t)b * S + j) * cin + c];
        }
        agg[(size_t)blockIdx.x * HC + e] = a * invc;
    }
}

// ---------------------------------------------------------------------------
// 3. bf16 WMMA GEMM:  C[M,N] = A[M,K] @ W[K,N] (+ bias), f32 in/out.
//    Tile 128x128, BK=64 (two WMMA k-steps/stage). 8 waves, each 32x64
//    (2 A-frags x 4 B-frags -> 16 WMMA per stage). W is staged TRANSPOSED
//    (Bt[n][k]) so every fragment load is contiguous (ds_load_b128).
//    K is zero-padded in LDS, so any K % 32 == 0 works (K=32 -> half stage).
// ---------------------------------------------------------------------------
#define BM  128
#define BN  128
#define BKK 64
#define LDA 72    // BKK + 8 pad; LDA*2 bytes = 144 (8B-aligned rows, bank-spread)
#define LDB 72

__global__ __launch_bounds__(256)
void mc_gemm_bf16_wmma_kernel(const float* __restrict__ A,
                              const float* __restrict__ W,
                              const float* __restrict__ bias,   // may be null
                              float* __restrict__ C,
                              int M, int N, int K)
{
    __shared__ __bf16 As[BM * LDA];   // [m][k]
    __shared__ __bf16 Bt[BN * LDB];   // [n][k]  (transposed!)

    const int tileM = blockIdx.y * BM;
    const int tileN = blockIdx.x * BN;

    const int lane  = threadIdx.x & 31;
    const int wid   = threadIdx.x >> 5;    // 0..7
    const int wm    = wid >> 1;            // 0..3  (32-row strip)
    const int wn    = wid & 1;             // 0..1  (64-col strip)
    const int lrow  = lane & 15;
    const int khalf = lane >> 4;           // 0 or 1

    v8f acc[2][4] = {};

    for (int k0 = 0; k0 < K; k0 += BKK) {
        // ---- stage A (BM x BKK), float4-vectorized, zero-pad M/K overrun ----
        for (int i = threadIdx.x; i < BM * BKK / 4; i += 256) {
            int r  = i >> 4;            // BKK/4 = 16 quads per row
            int c4 = (i & 15) << 2;
            int m  = tileM + r;
            f4 v = {0.0f, 0.0f, 0.0f, 0.0f};
            if (m < M && (k0 + c4) < K)              // K % 4 == 0 -> all-or-none
                v = *(const f4*)&A[(size_t)m * K + k0 + c4];
            bf4 o;
            #pragma unroll
            for (int u = 0; u < 4; ++u) o[u] = (__bf16)v[u];
            *(bf4*)&As[r * LDA + c4] = o;
        }
        // ---- stage W transposed: Bt[n][k], coalesced f4 reads along n ----
        for (int i = threadIdx.x; i < BKK * BN / 4; i += 256) {
            int r  = i >> 5;            // k row, BN/4 = 32 quads per row
            int c4 = (i & 31) << 2;     // n offset
            int kk = k0 + r;
            f4 v = {0.0f, 0.0f, 0.0f, 0.0f};
            if (kk < K) {
                int n = tileN + c4;
                if (n + 3 < N) {
                    v = *(const f4*)&W[(size_t)kk * N + n];
                } else {
                    #pragma unroll
                    for (int u = 0; u < 4; ++u)
                        if (n + u < N) v[u] = W[(size_t)kk * N + n + u];
                }
            }
            #pragma unroll
            for (int u = 0; u < 4; ++u)
                Bt[(c4 + u) * LDB + r] = (__bf16)v[u];
        }
        // hint next K-slab into caches (global_prefetch_b8)
        if (k0 + BKK < K) {
            int mp = tileM + (threadIdx.x >> 1);
            if (mp < M) __builtin_prefetch(&A[(size_t)mp * K + k0 + BKK], 0, 1);
            int kp = k0 + BKK + (threadIdx.x & 63);
            if (kp < K) __builtin_prefetch(&W[(size_t)kp * N + tileN], 0, 1);
        }
        __syncthreads();

        // ---- two WMMA k-steps per stage ----
        #pragma unroll
        for (int ks = 0; ks < BKK; ks += 32) {
            v16bf af[2], bfv[4];
            const int kbA = ks + khalf * 8;    // A: lanes 0-15 K+0.., 16-31 K+8..
            const int kbB = ks + khalf * 8;    // B: same k order as A
            #pragma unroll
            for (int i = 0; i < 2; ++i) {
                const __bf16* ap = &As[(wm * 32 + i * 16 + lrow) * LDA];
                #pragma unroll
                for (int e = 0; e < 8; ++e)  af[i][e] = ap[kbA + e];       // k=kb+e
                #pragma unroll
                for (int e = 8; e < 16; ++e) af[i][e] = ap[kbA + 8 + e];   // k=kb+16+(e-8)
            }
            #pragma unroll
            for (int j = 0; j < 4; ++j) {
                const __bf16* bp = &Bt[(wn * 64 + j * 16 + lrow) * LDB + kbB];
                #pragma unroll
                for (int e = 0; e < 16; ++e) bfv[j][e] = bp[e < 8 ? e : e + 8]; // k=kb+e / kb+16+(e-8)
            }
            #pragma unroll
            for (int i = 0; i < 2; ++i)
                #pragma unroll
                for (int j = 0; j < 4; ++j)
                    acc[i][j] = __builtin_amdgcn_wmma_f32_16x16x32_bf16(
                        false, af[i], false, bfv[j],
                        (short)0, acc[i][j], false, false);
        }
        __syncthreads();
    }

    // ---- epilogue: optional bias, guarded stores ----
    #pragma unroll
    for (int i = 0; i < 2; ++i) {
        #pragma unroll
        for (int j = 0; j < 4; ++j) {
            int col = tileN + wn * 64 + j * 16 + lrow;
            if (col >= N) continue;
            float bv = bias ? bias[col] : 0.0f;
            #pragma unroll
            for (int r = 0; r < 8; ++r) {
                int m = tileM + wm * 32 + i * 16 + khalf * 8 + r;
                if (m < M) C[(size_t)m * N + col] = acc[i][j][r] + bv;
            }
        }
    }
}

// ---------------------------------------------------------------------------
// 4. Fused BatchNorm (inference form) + LeakyReLU, in place.
// ---------------------------------------------------------------------------
__global__ void mc_bn_lrelu_kernel(float* __restrict__ x,
                                   const float* __restrict__ g,
                                   const float* __restrict__ b,
                                   int n, int Cch)
{
    int i = blockIdx.x * blockDim.x + threadIdx.x;
    if (i >= n) return;
    int c = i % Cch;
    float v = g[c] * (x[i] * BN_INV) + b[c];
    x[i] = lrelu(v);
}

// ---------------------------------------------------------------------------
// Host orchestration
// ---------------------------------------------------------------------------
extern "C" void kernel_launch(void* const* d_in, const int* in_sizes, int n_in,
                              void* d_out, int out_size, void* d_ws, size_t ws_size,
                              hipStream_t stream)
{
    (void)in_sizes; (void)n_in; (void)out_size; (void)ws_size;

    const float* points   = (const float*)d_in[0];
    const float* features = (const float*)d_in[1];
    const float* wb[4] = { (const float*)d_in[3],  (const float*)d_in[6],
                           (const float*)d_in[9],  (const float*)d_in[12] };
    const float* bb[4] = { (const float*)d_in[4],  (const float*)d_in[7],
                           (const float*)d_in[10], (const float*)d_in[13] };
    const float* wc[4] = { (const float*)d_in[5],  (const float*)d_in[8],
                           (const float*)d_in[11], (const float*)d_in[14] };
    const float* w11[3] = { (const float*)d_in[15], (const float*)d_in[17], (const float*)d_in[19] };
    const float* b11[3] = { (const float*)d_in[16], (const float*)d_in[18], (const float*)d_in[20] };
    const float* bng[8], *bnb[8];
    for (int k = 0; k < 8; ++k) {
        bng[k] = (const float*)d_in[21 + 2 * k];
        bnb[k] = (const float*)d_in[22 + 2 * k];
    }
    const float* d0w = (const float*)d_in[37];
    const float* d0b = (const float*)d_in[38];
    const float* d1w = (const float*)d_in[39];
    const float* d1b = (const float*)d_in[40];

    // ---- carve workspace ----
    char* wsp = (char*)d_ws;
    size_t off = 0;
    auto carve = [&](size_t bytes) -> void* {
        void* p = wsp + off;
        off += (bytes + 255) & ~(size_t)255;
        return p;
    };
    int*   idxb = (int*)  carve((size_t)16 * 1024 * 32 * sizeof(int));
    float* cntb = (float*)carve((size_t)16 * 1024 * sizeof(float));
    float* xa   = (float*)carve((size_t)16 * 1024 * 128 * sizeof(float));   // 8 MB
    float* xb   = (float*)carve((size_t)16 * 1024 * 128 * sizeof(float));   // 8 MB
    float* aggb = (float*)carve((size_t)16 * 256 * 4096 * sizeof(float));   // 64 MB

    const int   counts[5] = { 4096, 1024, 256, 64, 1 };
    const int   Ks[4]     = { 32, 32, 32, 64 };
    const int   feats[5]  = { 1, 128, 256, 512, 1024 };
    const float radii[4]  = { 0.1f, 0.2f, 0.4f, 3.4641016151377544f };

    auto gemm = [&](const float* A, const float* Wm, const float* bias,
                    float* Cm, int M, int N, int Kd) {
        dim3 g((unsigned)((N + BN - 1) / BN), (unsigned)((M + BM - 1) / BM));
        mc_gemm_bf16_wmma_kernel<<<g, 256, 0, stream>>>(A, Wm, bias, Cm, M, N, Kd);
    };

    const float* xsrc = features;
    int S = 4096, cin = 1;

    for (int l = 0; l < 4; ++l) {
        const int   Q    = counts[l + 1];
        const int   K    = Ks[l];
        const int   cout = feats[l + 1];
        const float r    = radii[l];

        mc_knn_kernel<<<(16 * Q + 127) / 128, 128, 0, stream>>>(
            points, S, Q, K, r * r, idxb, cntb);
        mc_agg_kernel<<<16 * Q, 256, 0, stream>>>(
            points, xsrc, cin, S, Q, K, 1.0f / r, wb[l], bb[l], idxb, cntb, aggb);
        gemm(aggb, wc[l], nullptr, xa, 16 * Q, cout, 32 * cin);

        if (l < 3) {
            int n = 16 * Q * cout;
            mc_bn_lrelu_kernel<<<(n + 255) / 256, 256, 0, stream>>>(
                xa, bng[2 * l], bnb[2 * l], n, cout);
            gemm(xa, w11[l], b11[l], xb, 16 * Q, cout, cout);
            mc_bn_lrelu_kernel<<<(n + 255) / 256, 256, 0, stream>>>(
                xb, bng[2 * l + 1], bnb[2 * l + 1], n, cout);
            xsrc = xb; S = Q; cin = cout;
        }
    }

    // ---- classifier tail: xa holds (16, 1024) ----
    int n = 16 * 1024;
    mc_bn_lrelu_kernel<<<(n + 255) / 256, 256, 0, stream>>>(xa, bng[6], bnb[6], n, 1024);
    gemm(xa, d0w, d0b, xb, 16, 1024, 1024);
    mc_bn_lrelu_kernel<<<(n + 255) / 256, 256, 0, stream>>>(xb, bng[7], bnb[7], n, 1024);
    gemm(xb, d1w, d1b, (float*)d_out, 16, 40, 1024);
}
